// TagSequenceModel_64802466562825
// MI455X (gfx1250) — matrix operand order repaired and checked by reading, not compiled, not run on hardware
//
#include <hip/hip_runtime.h>

typedef __attribute__((ext_vector_type(16))) _Float16 v16h;
typedef __attribute__((ext_vector_type(8)))  float    v8f;

#define TT        2048
#define NTAG      64
#define START_TAG 1
#define END_TAG   2
#define NEG_INF   (-10000.0f)
#define LOG2E     1.4426950408889634f
#define LN2       0.6931471805599453f

#define PSTR 72   // halves per p-row in LDS (padded: 16 lanes hit distinct bank groups)
#define USTR 68   // floats per unary-row in LDS (padded)

union HalfPack8 { _Float16 h[8]; uint4 q; };
union FragB     { v16h v; uint4 q[2]; };
union Float8    { float4 f4[2]; float f[8]; };

__launch_bounds__(32, 1)
__global__ void crf_forward_kernel(const float* __restrict__ unary,
                                   const float* __restrict__ trans,
                                   const int*   __restrict__ lengths,
                                   float*       __restrict__ out)
{
    __shared__ __align__(16) _Float16 p_lds[16 * PSTR];       // exp(alpha-m), [batch][tag] halves
    __shared__ __align__(16) float    u_lds[2][16 * USTR];    // unary double buffer, [batch][tag]

    const int lane = threadIdx.x;     // 0..31
    const int b    = lane & 15;       // batch column within tile
    const int hi   = lane >> 4;       // half-wave selector
    const int b0   = blockIdx.x * 16; // global batch base

    const int len_b = lengths[b0 + b];

    // ---------------- A fragments: E = exp(trans), f16, resident in VGPRs ----------------
    // 16-bit A 16x32 layout: lane row M = lane&15; per-VGPR K mapping per ISA table.
    v16h afrag[4][2];
    #pragma unroll
    for (int tl = 0; tl < 4; ++tl) {
        const int j = tl * 16 + b;                 // to-tag row
        #pragma unroll
        for (int kh = 0; kh < 2; ++kh) {
            v16h a;
            #pragma unroll
            for (int h = 0; h < 16; ++h) {
                const int v  = h >> 1, hl = h & 1;
                const int k  = kh * 32 + ((v < 4) ? 0 : 16) + hi * 8 + (v & 3) * 2 + hl;
                a[h] = (_Float16)__expf(trans[j * NTAG + k]);
            }
            afrag[tl][kh] = a;
        }
    }

    // ---------------- alpha state in D-tile layout: alph[tl][r] <-> tag 16*tl+8*hi+r ----
    float alph[4][8];
    #pragma unroll
    for (int tl = 0; tl < 4; ++tl)
        #pragma unroll
        for (int r = 0; r < 8; ++r) {
            const int tag = tl * 16 + hi * 8 + r;
            alph[tl][r] = (tag == START_TAG) ? 0.0f : NEG_INF;
        }

    // ---------------- preload unary[t=0] into LDS (coalesced b128) -----------------------
    #pragma unroll
    for (int i = 0; i < 8; ++i) {
        const int idx = i * 32 + lane;
        const int bl = idx >> 4, v4 = idx & 15;
        const float4 u = *(const float4*)(unary + ((size_t)(b0 + bl) * TT) * NTAG + v4 * 4);
        *(float4*)&u_lds[0][bl * USTR + v4 * 4] = u;
    }

    int cur = 0;
    for (int t = 0; t < TT; ++t) {
        // ---- issue next-step unary loads early (clamped: straight-line body, no tail dup)
        const int tn = (t + 1 < TT) ? (t + 1) : (TT - 1);
        float4 u4[8];
        #pragma unroll
        for (int i = 0; i < 8; ++i) {
            const int idx = i * 32 + lane;
            const int bl = idx >> 4, v4 = idx & 15;
            u4[i] = *(const float4*)(unary + ((size_t)(b0 + bl) * TT + tn) * NTAG + v4 * 4);
        }
        {
            const int tp = (t + 2 < TT) ? (t + 2) : (TT - 1);
            __builtin_prefetch(unary + ((size_t)(b0 + (lane >> 1)) * TT + tp) * NTAG
                                     + (lane & 1) * 32, 0, 1);   // -> global_prefetch_b8
        }

        // ---- per-batch-column max over all 64 tags -----------------------------------
        float m = NEG_INF;
        #pragma unroll
        for (int tl = 0; tl < 4; ++tl)
            #pragma unroll
            for (int r = 0; r < 8; ++r) m = fmaxf(m, alph[tl][r]);
        m = fmaxf(m, __shfl_xor(m, 16, 32));

        // ---- P = exp(alpha - m) as f16 -> LDS rows [batch][tag] ----------------------
        #pragma unroll
        for (int tl = 0; tl < 4; ++tl) {
            HalfPack8 pk;
            #pragma unroll
            for (int r = 0; r < 8; ++r)
                pk.h[r] = (_Float16)__builtin_amdgcn_exp2f((alph[tl][r] - m) * LOG2E);
            *(uint4*)&p_lds[b * PSTR + tl * 16 + hi * 8] = pk.q;   // ds_store_b128
        }
        // same-wave LDS is in-order: no barrier needed before the fragment reads

        // ---- B fragments: K-halves 8*hi..+7 and 16+8*hi..+7 per lane (table order) ---
        FragB bf[2];
        #pragma unroll
        for (int kh = 0; kh < 2; ++kh) {
            const _Float16* src = &p_lds[b * PSTR + kh * 32 + hi * 8];
            bf[kh].q[0] = *(const uint4*)(src);       // ds_load_b128
            bf[kh].q[1] = *(const uint4*)(src + 16);  // ds_load_b128
        }

        // ---- 4 independent tile chains: all K0 WMMAs, then all K1 WMMAs --------------
        v8f acc[4];
        #pragma unroll
        for (int tl = 0; tl < 4; ++tl) {
            v8f z = {0.f, 0.f, 0.f, 0.f, 0.f, 0.f, 0.f, 0.f};
            acc[tl] = __builtin_amdgcn_wmma_f32_16x16x32_f16(
                          false, afrag[tl][0], false, bf[0].v, (short)0, z, false, false);
        }
        #pragma unroll
        for (int tl = 0; tl < 4; ++tl)
            acc[tl] = __builtin_amdgcn_wmma_f32_16x16x32_f16(
                          false, afrag[tl][1], false, bf[1].v, (short)0, acc[tl], false, false);

        // ---- unary for this step (unconditional b128 pairs) --------------------------
        Float8 uu[4];
        #pragma unroll
        for (int tl = 0; tl < 4; ++tl) {
            const float* up = &u_lds[cur][b * USTR + tl * 16 + hi * 8];
            uu[tl].f4[0] = *(const float4*)(up);
            uu[tl].f4[1] = *(const float4*)(up + 4);
        }

        // ---- alpha update, branchless length mask (v_bfi_b32) ------------------------
        const unsigned keep = (t < len_b) ? 0xFFFFFFFFu : 0u;
        #pragma unroll
        for (int tl = 0; tl < 4; ++tl)
            #pragma unroll
            for (int r = 0; r < 8; ++r) {
                const float na = m + __builtin_amdgcn_logf(acc[tl][r]) * LN2 + uu[tl].f[r];
                const unsigned nb = __float_as_uint(na);
                const unsigned ob = __float_as_uint(alph[tl][r]);
                alph[tl][r] = __uint_as_float((keep & nb) | (~keep & ob));
            }

        // ---- stage next unary into the other LDS buffer ------------------------------
        const int nxt = cur ^ 1;
        #pragma unroll
        for (int i = 0; i < 8; ++i) {
            const int idx = i * 32 + lane;
            const int bl = idx >> 4, v4 = idx & 15;
            *(float4*)&u_lds[nxt][bl * USTR + v4 * 4] = u4[i];
        }
        cur = nxt;
    }

    // ---------------- terminal: alpha + trans[EOS][k], logsumexp over k ----------------
    float term[4][8];
    float mx = NEG_INF;
    #pragma unroll
    for (int tl = 0; tl < 4; ++tl)
        #pragma unroll
        for (int r = 0; r < 8; ++r) {
            const int tag = tl * 16 + hi * 8 + r;
            term[tl][r] = alph[tl][r] + trans[END_TAG * NTAG + tag];
            mx = fmaxf(mx, term[tl][r]);
        }
    mx = fmaxf(mx, __shfl_xor(mx, 16, 32));
    float s = 0.f;
    #pragma unroll
    for (int tl = 0; tl < 4; ++tl)
        #pragma unroll
        for (int r = 0; r < 8; ++r)
            s += __builtin_amdgcn_exp2f((term[tl][r] - mx) * LOG2E);
    s += __shfl_xor(s, 16, 32);
    if (hi == 0)
        out[b0 + b] = mx + __builtin_amdgcn_logf(s) * LN2;
}

extern "C" void kernel_launch(void* const* d_in, const int* in_sizes, int n_in,
                              void* d_out, int out_size, void* d_ws, size_t ws_size,
                              hipStream_t stream) {
    const float* unary   = (const float*)d_in[0];   // [B, T, N] f32
    const float* trans   = (const float*)d_in[1];   // [N, N]    f32
    const int*   lengths = (const int*)  d_in[2];   // [B]       i32
    float*       out     = (float*)d_out;           // [B]       f32

    const int B = in_sizes[2];                      // 256
    dim3 grid(B / 16), block(32);                   // one wave32 per 16-batch tile
    hipLaunchKernelGGL(crf_forward_kernel, grid, block, 0, stream,
                       unary, trans, lengths, out);
}
